// RSSM_61753039782053
// MI455X (gfx1250) — matrix operand order repaired and checked by reading, not compiled, not run on hardware
//
#include <hip/hip_runtime.h>
#include <hip/hip_bf16.h>
#include <stddef.h>
#include <stdint.h>

// ---------------------------------------------------------------------------
// RSSM rollout for MI455X (gfx1250).
// B=256, T=64, E=1024, A=32, D=1024, S=256, N=1024, K(ens)=5.
// ~390 GFLOP total; all weights ~24MB bf16 -> resident in 192MB L2, so each
// of the 63 sequential steps is pure matrix-core work. Strategy:
//   - one-time fp32->bf16 weight repack (transpose W_ih/W_hh, pack W_ens)
//   - bf16 WMMA GEMMs (v_wmma_f32_16x16x32_bf16), fp32 accumulate
//   - A-tile staging via GLOBAL_LOAD_ASYNC_TO_LDS_B128 (ASYNCcnt) when the
//     toolchain exposes the builtin; VGPR-path fallback otherwise
//   - concat inputs handled as two accumulating GEMMs (C-load epilogue)
//   - bias / ELU / dual fp32+bf16 stores fused into GEMM epilogue
//   - GRU gates + softplus/noise fused pointwise kernels, writing directly
//     into the strided [B, T-1, 4864] output.
// ---------------------------------------------------------------------------

typedef __attribute__((ext_vector_type(16))) __bf16 v16bf;
typedef __attribute__((ext_vector_type(8)))  float  v8f;
typedef __attribute__((ext_vector_type(4)))  int    v4i_t;

#define AS1 __attribute__((address_space(1)))
#define AS3 __attribute__((address_space(3)))
typedef AS1 v4i_t* gptr128;
typedef AS3 v4i_t* lptr128;

#if __has_builtin(__builtin_amdgcn_global_load_async_to_lds_b128)
#define HAVE_ASYNC_LDS 1
#else
#define HAVE_ASYNC_LDS 0
#endif

union FragBF {
    unsigned int u[8];
    v16bf v;
};

__device__ __forceinline__ unsigned short f2bf(float x) {
    unsigned int u = __float_as_uint(x);
    u += 0x7FFFu + ((u >> 16) & 1u);   // round-to-nearest-even
    return (unsigned short)(u >> 16);
}

__device__ __forceinline__ void wait_async_zero() {
#if HAVE_ASYNC_LDS
#if __has_builtin(__builtin_amdgcn_s_wait_asynccnt)
    __builtin_amdgcn_s_wait_asynccnt(0);
#else
    asm volatile("s_wait_asynccnt 0" ::: "memory");
#endif
#endif
}

// ---------------------------------------------------------------------------
// Generic bf16 GEMM: C[M,N] = act( A[M,K](bf16) * B[K,N](bf16) + bias + Cin )
// Block tile 64x128, 8 waves (2x4), each wave 32x32 (2x2 WMMA 16x16x32 tiles).
// ACT: 0=none, 1=elu.  LOADC: add fp32 Cin.  OUTF/OUTB: fp32 / bf16 stores.
// ---------------------------------------------------------------------------
template <int ACT, int LOADC, int OUTF, int OUTB>
__global__ __launch_bounds__(256) void gemm_bf16_kernel(
    const unsigned short* __restrict__ A, int lda,
    const unsigned short* __restrict__ Bm, int ldb,
    const float* __restrict__ bias,
    const float* __restrict__ Cin, int ldcin,
    float* __restrict__ Cf, int ldcf,
    unsigned short* __restrict__ Cb, int ldcb,
    int M, int N, int K)
{
    // As stride 40 ushorts = 80B: 16B-aligned rows (async b128 requirement),
    // and row*20 dwords mod 64 banks distinct for 16 consecutive rows.
    __shared__ unsigned short As[64 * 40];    // row-major A tile
    __shared__ unsigned short Bs[128 * 34];   // K-transposed: Bs[col*34 + k]
    (void)M;

    const int tid  = threadIdx.x;
    const int lane = tid & 31;
    const int w    = tid >> 5;
    const int wm   = w >> 2;   // 0..1  (rows of 32)
    const int wn   = w & 3;    // 0..3  (cols of 32)
    const int n0   = blockIdx.x * 128;
    const int m0   = blockIdx.y * 64;

    v8f acc[2][2] = {};

    // staging coordinates
    const int ar = tid >> 2;          // 0..63  A row
    const int ac = (tid & 3) * 8;     // 0/8/16/24 A col chunk (8 bf16 = 16B)
    const int bk = tid >> 3;          // 0..31  B row (K)
    const int bc = (tid & 7) * 16;    // B col chunk (16 bf16 = 32B)

    // fragment coordinates (contiguous k = 16*(lane/16) + s for both A and B, wave32)
    const int fr  = lane & 15;
    const int akb = (lane >> 4) * 16;  // A K-base: lanes 0-15 -> 0, 16-31 -> 16
    const int bkb = (lane >> 4) * 16;  // B K-base: lanes 0-15 -> 0, 16-31 -> 16

    for (int k0 = 0; k0 < K; k0 += 32) {
        // ---- stage A tile (64x32): direct global->LDS async copy if avail ----
#if HAVE_ASYNC_LDS
        __builtin_amdgcn_global_load_async_to_lds_b128(
            (gptr128)(uintptr_t)(A + (size_t)(m0 + ar) * lda + k0 + ac),
            (lptr128)(uintptr_t)&As[ar * 40 + ac], 0, 0);
#else
        {
            const uint4 va = *(const uint4*)(A + (size_t)(m0 + ar) * lda + k0 + ac);
            unsigned int* dst = (unsigned int*)&As[ar * 40 + ac];
            dst[0] = va.x; dst[1] = va.y; dst[2] = va.z; dst[3] = va.w;
        }
#endif
        // ---- stage B tile (32x128), transposed into Bs[col][k] ----
        {
            const unsigned short* src = Bm + (size_t)(k0 + bk) * ldb + n0 + bc;
            uint4 v0 = *(const uint4*)(src);
            uint4 v1 = *(const uint4*)(src + 8);
            unsigned short tmp[16];
            *(uint4*)(tmp)     = v0;
            *(uint4*)(tmp + 8) = v1;
#pragma unroll
            for (int c = 0; c < 16; ++c)
                Bs[(bc + c) * 34 + bk] = tmp[c];
        }
        // prefetch next K tile of B (largest stream) into cache
        if (k0 + 32 < K)
            __builtin_prefetch(Bm + (size_t)(k0 + 32 + bk) * ldb + n0 + bc, 0, 3);

        wait_async_zero();
        __syncthreads();

        // ---- build fragments ----
        FragBF af[2], bf[2];
#pragma unroll
        for (int h = 0; h < 2; ++h) {
            const unsigned short* ap = &As[(wm * 32 + h * 16 + fr) * 40];
#pragma unroll
            for (int i = 0; i < 8; ++i) {
                const int kk = akb + 2 * i;
                af[h].u[i] = *(const unsigned int*)(ap + kk);
            }
            const unsigned short* bp = &Bs[(wn * 32 + h * 16 + fr) * 34 + bkb];
#pragma unroll
            for (int i = 0; i < 8; ++i)
                bf[h].u[i] = *(const unsigned int*)(bp + 2 * i);
        }

        // ---- 2x2 WMMA ----
#pragma unroll
        for (int ti = 0; ti < 2; ++ti)
#pragma unroll
            for (int tj = 0; tj < 2; ++tj)
                acc[ti][tj] = __builtin_amdgcn_wmma_f32_16x16x32_bf16(
                    false, af[ti].v, false, bf[tj].v,
                    (short)0, acc[ti][tj], false, false);

        __syncthreads();
    }

    // ---- epilogue ----
#pragma unroll
    for (int ti = 0; ti < 2; ++ti) {
        const int rbase = m0 + wm * 32 + ti * 16 + (lane >> 4) * 8;
#pragma unroll
        for (int tj = 0; tj < 2; ++tj) {
            const int c = n0 + wn * 32 + tj * 16 + (lane & 15);
            const float bv = bias ? bias[c] : 0.0f;
#pragma unroll
            for (int v = 0; v < 8; ++v) {
                const int r = rbase + v;
                float x = acc[ti][tj][v] + bv;
                if (LOADC) x += Cin[(size_t)r * ldcin + c];
                if (ACT == 1) x = (x > 0.0f) ? x : (expf(x) - 1.0f);
                if (OUTF) Cf[(size_t)r * ldcf + c] = x;
                if (OUTB) Cb[(size_t)r * ldcb + c] = f2bf(x);
            }
        }
    }
}

// ---------------------------------------------------------------------------
// Weight repack kernels (once per launch)
// ---------------------------------------------------------------------------
__global__ void conv_bf16_kernel(const float* __restrict__ src,
                                 unsigned short* __restrict__ dst, int n) {
    int i = blockIdx.x * blockDim.x + threadIdx.x;
    if (i < n) dst[i] = f2bf(src[i]);
}

// src [rows, cols] row-major  ->  dst [cols, rows] row-major (i.e. src^T)
__global__ void conv_transpose_kernel(const float* __restrict__ src,
                                      unsigned short* __restrict__ dst,
                                      int rows, int cols) {
    int i = blockIdx.x * blockDim.x + threadIdx.x;
    if (i >= rows * cols) return;
    int d = i / rows;   // 0..cols-1
    int j = i % rows;   // 0..rows-1
    dst[(size_t)d * rows + j] = f2bf(src[(size_t)j * cols + d]);
}

// W_ens [5,1024,256] -> packed B [1024, 1280] with col j = k*256 + s
__global__ void conv_ens_kernel(const float* __restrict__ src,
                                unsigned short* __restrict__ dst) {
    int i = blockIdx.x * blockDim.x + threadIdx.x;
    if (i >= 1024 * 1280) return;
    int d = i / 1280;
    int j = i % 1280;
    int k = j >> 8;
    int s = j & 255;
    dst[i] = f2bf(src[((size_t)k * 1024 + d) * 256 + s]);
}

// ---------------------------------------------------------------------------
// Per-step pointwise kernels
// ---------------------------------------------------------------------------
// builds stoch*nt (bf16), action*nt (bf16), deter*nt (fp32 + bf16), obs (bf16)
__global__ void prep_step_kernel(const float* __restrict__ stoch_state,
                                 const float* __restrict__ deter_state,
                                 const float* __restrict__ action,
                                 const float* __restrict__ nonterms,
                                 const float* __restrict__ obs,
                                 int t,
                                 unsigned short* __restrict__ stochnt_bf,
                                 unsigned short* __restrict__ actnt_bf,
                                 unsigned short* __restrict__ h_bf,
                                 float* __restrict__ h_f32,
                                 unsigned short* __restrict__ obs_bf) {
    const int PER_B = 256 + 32 + 1024 + 1024;  // 2336
    int idx = blockIdx.x * blockDim.x + threadIdx.x;
    if (idx >= 256 * PER_B) return;
    int b = idx / PER_B;
    int j = idx % PER_B;
    float nt = nonterms[b * 64 + t];
    if (j < 256) {
        stochnt_bf[b * 256 + j] = f2bf(stoch_state[b * 256 + j] * nt);
    } else if (j < 288) {
        int a1 = j - 256;
        actnt_bf[b * 32 + a1] = f2bf(action[((size_t)b * 64 + t) * 32 + a1] * nt);
    } else if (j < 1312) {
        int d = j - 288;
        float h = deter_state[(size_t)b * 1024 + d] * nt;
        h_f32[(size_t)b * 1024 + d] = h;
        h_bf[(size_t)b * 1024 + d]  = f2bf(h);
    } else {
        int e = j - 1312;
        obs_bf[(size_t)b * 1024 + e] = f2bf(obs[((size_t)b * 64 + t) * 1024 + e]);
    }
}

// GRU gate math; writes deter_new (fp32 state + bf16) and into out[.., 768/2560]
__global__ void gru_fuse_kernel(const float* __restrict__ gi,
                                const float* __restrict__ gh,
                                const float* __restrict__ h,
                                float* __restrict__ deter_state,
                                unsigned short* __restrict__ deter_bf,
                                float* __restrict__ out, int t) {
    int idx = blockIdx.x * blockDim.x + threadIdx.x;
    if (idx >= 256 * 1024) return;
    int b = idx >> 10;
    int d = idx & 1023;
    size_t gb = (size_t)b * 3072;
    float ir = gi[gb + d],        hr = gh[gb + d];
    float iz = gi[gb + 1024 + d], hz = gh[gb + 1024 + d];
    float ig = gi[gb + 2048 + d], hg = gh[gb + 2048 + d];
    float r = 1.0f / (1.0f + expf(-(ir + hr)));
    float z = 1.0f / (1.0f + expf(-(iz + hz)));
    float n = tanhf(ig + r * hg);
    float dn = (1.0f - z) * n + z * h[idx];
    deter_state[idx] = dn;
    deter_bf[idx]    = f2bf(dn);
    size_t base = ((size_t)b * 63 + t) * 4864;
    out[base + 768 + d]  = dn;   // prior_cat tail
    out[base + 2560 + d] = dn;   // post_cat tail
}

// softplus + noise; writes pm/ps/prior and qm/qs/post into out; updates stoch
__global__ void dist_fuse_kernel(const float* __restrict__ p2,
                                 const float* __restrict__ q2,
                                 const float* __restrict__ eps_p,
                                 const float* __restrict__ eps_q,
                                 float* __restrict__ stoch_state,
                                 float* __restrict__ out, int t) {
    int idx = blockIdx.x * blockDim.x + threadIdx.x;
    if (idx >= 256 * 256) return;
    int b = idx >> 8;
    int s = idx & 255;
    size_t pb = (size_t)b * 512;
    float pm  = p2[pb + s];
    float psr = p2[pb + 256 + s];
    float ps  = ((psr > 20.0f) ? psr : log1pf(expf(psr))) + 0.1f;
    float pr  = pm + ps * eps_p[((size_t)t * 256 + b) * 256 + s];
    float qm  = q2[pb + s];
    float qsr = q2[pb + 256 + s];
    float qs  = ((qsr > 20.0f) ? qsr : log1pf(expf(qsr))) + 0.1f;
    float po  = qm + qs * eps_q[((size_t)t * 256 + b) * 256 + s];
    size_t base = ((size_t)b * 63 + t) * 4864;
    out[base + s]        = pm;
    out[base + 256 + s]  = ps;
    out[base + 512 + s]  = pr;
    out[base + 1792 + s] = qm;
    out[base + 2048 + s] = qs;
    out[base + 2304 + s] = po;
    stoch_state[(size_t)b * 256 + s] = po;
}

// ---------------------------------------------------------------------------
// Host side
// ---------------------------------------------------------------------------
extern "C" void kernel_launch(void* const* d_in, const int* in_sizes, int n_in,
                              void* d_out, int out_size, void* d_ws, size_t ws_size,
                              hipStream_t stream) {
    (void)in_sizes; (void)n_in; (void)out_size; (void)ws_size;

    const float* obs       = (const float*)d_in[1];
    const float* action    = (const float*)d_in[2];
    const float* nonterms  = (const float*)d_in[3];
    const float* init_det  = (const float*)d_in[4];
    const float* init_sto  = (const float*)d_in[5];
    const float* eps_p     = (const float*)d_in[6];
    const float* eps_q     = (const float*)d_in[7];
    const float* W_embed   = (const float*)d_in[8];
    const float* b_embed   = (const float*)d_in[9];
    const float* W_ens     = (const float*)d_in[10];
    const float* b_ens     = (const float*)d_in[11];
    const float* W_ih      = (const float*)d_in[12];
    const float* W_hh      = (const float*)d_in[13];
    const float* b_ih      = (const float*)d_in[14];
    const float* b_hh      = (const float*)d_in[15];
    const float* Wp1       = (const float*)d_in[16];
    const float* bp1       = (const float*)d_in[17];
    const float* Wp2       = (const float*)d_in[18];
    const float* bp2       = (const float*)d_in[19];
    const float* Wq1       = (const float*)d_in[20];
    const float* bq1       = (const float*)d_in[21];
    const float* Wq2       = (const float*)d_in[22];
    const float* bq2       = (const float*)d_in[23];
    float* out = (float*)d_out;

    // --- scratch bump allocator (~40MB total) ---
    size_t off = 0;
    auto alloc = [&](size_t bytes) -> void* {
        off = (off + 255) & ~(size_t)255;
        void* q = (char*)d_ws + off;
        off += bytes;
        return q;
    };
    typedef unsigned short us;
    us* Wemb_bf  = (us*)alloc((size_t)288 * 1024 * 2);
    us* Wens_bf  = (us*)alloc((size_t)1024 * 1280 * 2);
    us* WihT_bf  = (us*)alloc((size_t)1024 * 3072 * 2);
    us* WhhT_bf  = (us*)alloc((size_t)1024 * 3072 * 2);
    us* Wp1_bf   = (us*)alloc((size_t)1024 * 1024 * 2);
    us* Wq1_bf   = (us*)alloc((size_t)2048 * 1024 * 2);
    us* Wp2_bf   = (us*)alloc((size_t)1024 * 512 * 2);
    us* Wq2_bf   = (us*)alloc((size_t)1024 * 512 * 2);
    us* stochnt_bf = (us*)alloc((size_t)256 * 256 * 2);
    us* actnt_bf   = (us*)alloc((size_t)256 * 32 * 2);
    us* h_bf       = (us*)alloc((size_t)256 * 1024 * 2);
    us* obs_bf     = (us*)alloc((size_t)256 * 1024 * 2);
    us* sa_bf      = (us*)alloc((size_t)256 * 1024 * 2);
    us* deter_bf   = (us*)alloc((size_t)256 * 1024 * 2);
    us* p1_bf      = (us*)alloc((size_t)256 * 1024 * 2);
    us* q1_bf      = (us*)alloc((size_t)256 * 1024 * 2);
    float* csa         = (float*)alloc((size_t)256 * 1024 * 4);
    float* cq          = (float*)alloc((size_t)256 * 1024 * 4);
    float* gi          = (float*)alloc((size_t)256 * 3072 * 4);
    float* gh          = (float*)alloc((size_t)256 * 3072 * 4);
    float* h_f32       = (float*)alloc((size_t)256 * 1024 * 4);
    float* deter_state = (float*)alloc((size_t)256 * 1024 * 4);
    float* stoch_state = (float*)alloc((size_t)256 * 256 * 4);
    float* p2          = (float*)alloc((size_t)256 * 512 * 4);
    float* q2          = (float*)alloc((size_t)256 * 512 * 4);

    // --- one-time weight repack to bf16 ---
    auto cv = [&](const float* s, us* d, int n) {
        conv_bf16_kernel<<<(n + 255) / 256, 256, 0, stream>>>(s, d, n);
    };
    cv(W_embed, Wemb_bf, 288 * 1024);
    conv_ens_kernel<<<(1024 * 1280 + 255) / 256, 256, 0, stream>>>(W_ens, Wens_bf);
    conv_transpose_kernel<<<(3072 * 1024 + 255) / 256, 256, 0, stream>>>(W_ih, WihT_bf, 3072, 1024);
    conv_transpose_kernel<<<(3072 * 1024 + 255) / 256, 256, 0, stream>>>(W_hh, WhhT_bf, 3072, 1024);
    cv(Wp1, Wp1_bf, 1024 * 1024);
    cv(Wq1, Wq1_bf, 2048 * 1024);
    cv(Wp2, Wp2_bf, 1024 * 512);
    cv(Wq2, Wq2_bf, 1024 * 512);

    (void)hipMemcpyAsync(deter_state, init_det, (size_t)256 * 1024 * 4,
                         hipMemcpyDeviceToDevice, stream);
    (void)hipMemcpyAsync(stoch_state, init_sto, (size_t)256 * 256 * 4,
                         hipMemcpyDeviceToDevice, stream);

    const int LDOUT = 63 * 4864;

    for (int t = 0; t < 63; ++t) {
        prep_step_kernel<<<(256 * 2336 + 255) / 256, 256, 0, stream>>>(
            stoch_state, deter_state, action, nonterms, obs, t,
            stochnt_bf, actnt_bf, h_bf, h_f32, obs_bf);

        // sa = elu([stoch*nt, act*nt] @ W_embed + b) as two accumulating GEMMs
        gemm_bf16_kernel<0, 0, 1, 0><<<dim3(8, 4), 256, 0, stream>>>(
            stochnt_bf, 256, Wemb_bf, 1024, b_embed,
            nullptr, 0, csa, 1024, nullptr, 0, 256, 1024, 256);
        gemm_bf16_kernel<1, 1, 0, 1><<<dim3(8, 4), 256, 0, stream>>>(
            actnt_bf, 32, Wemb_bf + (size_t)256 * 1024, 1024, nullptr,
            csa, 1024, nullptr, 0, sa_bf, 1024, 256, 1024, 32);

        // ensemble preds -> directly into out[..., 3584:4864]
        gemm_bf16_kernel<0, 0, 1, 0><<<dim3(10, 4), 256, 0, stream>>>(
            sa_bf, 1024, Wens_bf, 1280, b_ens,
            nullptr, 0, out + (size_t)t * 4864 + 3584, LDOUT,
            nullptr, 0, 256, 1280, 1024);

        // GRU input/hidden GEMMs
        gemm_bf16_kernel<0, 0, 1, 0><<<dim3(24, 4), 256, 0, stream>>>(
            sa_bf, 1024, WihT_bf, 3072, b_ih,
            nullptr, 0, gi, 3072, nullptr, 0, 256, 3072, 1024);
        gemm_bf16_kernel<0, 0, 1, 0><<<dim3(24, 4), 256, 0, stream>>>(
            h_bf, 1024, WhhT_bf, 3072, b_hh,
            nullptr, 0, gh, 3072, nullptr, 0, 256, 3072, 1024);

        gru_fuse_kernel<<<(256 * 1024 + 255) / 256, 256, 0, stream>>>(
            gi, gh, h_f32, deter_state, deter_bf, out, t);

        // prior MLP layer 1
        gemm_bf16_kernel<1, 0, 0, 1><<<dim3(8, 4), 256, 0, stream>>>(
            deter_bf, 1024, Wp1_bf, 1024, bp1,
            nullptr, 0, nullptr, 0, p1_bf, 1024, 256, 1024, 1024);
        // posterior MLP layer 1: [deter, obs] as two accumulating GEMMs
        gemm_bf16_kernel<0, 0, 1, 0><<<dim3(8, 4), 256, 0, stream>>>(
            deter_bf, 1024, Wq1_bf, 1024, bq1,
            nullptr, 0, cq, 1024, nullptr, 0, 256, 1024, 1024);
        gemm_bf16_kernel<1, 1, 0, 1><<<dim3(8, 4), 256, 0, stream>>>(
            obs_bf, 1024, Wq1_bf + (size_t)1024 * 1024, 1024, nullptr,
            cq, 1024, nullptr, 0, q1_bf, 1024, 256, 1024, 1024);

        // layer 2s
        gemm_bf16_kernel<0, 0, 1, 0><<<dim3(4, 4), 256, 0, stream>>>(
            p1_bf, 1024, Wp2_bf, 512, bp2,
            nullptr, 0, p2, 512, nullptr, 0, 256, 512, 1024);
        gemm_bf16_kernel<0, 0, 1, 0><<<dim3(4, 4), 256, 0, stream>>>(
            q1_bf, 1024, Wq2_bf, 512, bq2,
            nullptr, 0, q2, 512, nullptr, 0, 256, 512, 1024);

        dist_fuse_kernel<<<(256 * 256 + 255) / 256, 256, 0, stream>>>(
            p2, q2, eps_p, eps_q, stoch_state, out, t);
    }
}
